// NonLocalBlock_21320217657912
// MI455X (gfx1250) — compile-verified, matrix operand order checked by reading, not run on hardware
//
#include <hip/hip_runtime.h>
#include <hip/hip_bf16.h>
#include <stdint.h>

// NonLocalBlock for MI455X (gfx1250, wave32, WMMA bf16 16x16x32).
// Sizes fixed by the reference: B=4, C=512, H=W=64 -> N=4096.
#define CC 512
#define NN 4096
#define NB 4

typedef __attribute__((ext_vector_type(16))) __bf16 v16bf;
typedef __attribute__((ext_vector_type(8)))  float  v8f;

union Frag {
  uint4  q[2];   // two 16B runs of 8 bf16 each (matches CDNA5 A/B 16-bit layout)
  v16bf  v;
};

__device__ __forceinline__ unsigned short f2bf(float f) {
  union { float f; uint32_t u; } c; c.f = f;
  uint32_t u = c.u;
  u += 0x7FFFu + ((u >> 16) & 1u);     // round-to-nearest-even
  return (unsigned short)(u >> 16);
}

__device__ __forceinline__ v8f vzero() {
  v8f z;
#pragma unroll
  for (int i = 0; i < 8; i++) z[i] = 0.0f;
  return z;
}

__device__ __forceinline__ v8f wmma_bf16(const Frag& a, const Frag& b, v8f c) {
  // (neg_a, A, neg_b, B, c_mod, C, reuse_a, reuse_b)
  return __builtin_amdgcn_wmma_f32_16x16x32_bf16(false, a.v, false, b.v,
                                                 (short)0, c, false, false);
}

// ---------------------------------------------------------------------------
// conv1x1 as GEMM: Y = W(512x512) @ X(512x4096) + bias, per batch.
// IN_BF16: 0 -> X is fp32 (C,N); 1 -> X is bf16 (C,N)
// MODE:    0 -> store bf16 (C,N); 1 -> store bf16 transposed (N,C);
//          2 -> store fp32 (C,N) with residual add (final layer)
// Tile: 128(M) x 128(N) x 32(K), 256 threads = 8 waves, each wave 2x4 WMMA tiles.
// ---------------------------------------------------------------------------
template<int IN_BF16, int MODE>
__global__ __launch_bounds__(256) void conv_gemm(
    const float* __restrict__ W, const float* __restrict__ bias,
    const void* __restrict__ Xv,
    unsigned short* __restrict__ Yb, float* __restrict__ Yf,
    const float* __restrict__ resid)
{
  constexpr int AST = 40;  // padded LDS stride (bf16 elems): 80B keeps 16B runs aligned, no bank clash
  __shared__ __align__(16) unsigned short Abuf[128][AST]; // W tile,  [m][k]
  __shared__ __align__(16) unsigned short Bbuf[128][AST]; // X tile,  [n][k]

  const int tid  = threadIdx.x;
  const int wv   = tid >> 5, lane = tid & 31;
  const int hf   = lane >> 4, ln = lane & 15;
  const int wm   = wv & 3;      // 4 wave-rows  * 32 rows
  const int wn   = wv >> 2;     // 2 wave-cols  * 64 cols
  const int obase = blockIdx.y * 128;
  const int nbase = blockIdx.x * 128;
  const int b     = blockIdx.z;

  const float*          Xf = (const float*)Xv          + (size_t)b * CC * NN;
  const unsigned short* Xh = (const unsigned short*)Xv + (size_t)b * CC * NN;

  v8f acc[2][4];
#pragma unroll
  for (int i = 0; i < 2; i++)
#pragma unroll
    for (int j = 0; j < 4; j++) acc[i][j] = vzero();

  for (int k0 = 0; k0 < CC; k0 += 32) {
    // ---- stage A: W tile (fp32 -> bf16), coalesced along K ----
    {
      const int row = tid >> 1, cg = (tid & 1) * 16;
      const float* src = W + (size_t)(obase + row) * CC + k0 + cg;
#pragma unroll
      for (int i = 0; i < 4; i++) {
        float4 v4 = *(const float4*)(src + 4 * i);
        uint2 p;
        p.x = (uint32_t)f2bf(v4.x) | ((uint32_t)f2bf(v4.y) << 16);
        p.y = (uint32_t)f2bf(v4.z) | ((uint32_t)f2bf(v4.w) << 16);
        *(uint2*)&Abuf[row][cg + 4 * i] = p;
      }
    }
    // ---- stage B: X tile -> Bbuf[n][k] (transpose into K-contiguous rows) ----
    {
      const int kk = tid >> 3, cg = (tid & 7) * 16;
      if constexpr (!IN_BF16) {
        const float* src = Xf + (size_t)(k0 + kk) * NN + nbase + cg;
#pragma unroll
        for (int i = 0; i < 4; i++) {
          float4 v4 = *(const float4*)(src + 4 * i);
          Bbuf[cg + 4 * i + 0][kk] = f2bf(v4.x);
          Bbuf[cg + 4 * i + 1][kk] = f2bf(v4.y);
          Bbuf[cg + 4 * i + 2][kk] = f2bf(v4.z);
          Bbuf[cg + 4 * i + 3][kk] = f2bf(v4.w);
        }
      } else {
        const unsigned short* src = Xh + (size_t)(k0 + kk) * NN + nbase + cg;
#pragma unroll
        for (int i = 0; i < 2; i++) {
          uint4 v4 = *(const uint4*)(src + 8 * i);
          const unsigned short* e = (const unsigned short*)&v4;
#pragma unroll
          for (int j = 0; j < 8; j++) Bbuf[cg + 8 * i + j][kk] = e[j];
        }
      }
    }
    __syncthreads();

    // ---- compute: 8 WMMA per wave per K-tile ----
    Frag af[2], bf4[4];
#pragma unroll
    for (int i = 0; i < 2; i++) {
      const int row = wm * 32 + i * 16 + ln;
      af[i].q[0] = *(const uint4*)&Abuf[row][hf * 8];
      af[i].q[1] = *(const uint4*)&Abuf[row][16 + hf * 8];
    }
#pragma unroll
    for (int j = 0; j < 4; j++) {
      const int col = wn * 64 + j * 16 + ln;
      bf4[j].q[0] = *(const uint4*)&Bbuf[col][hf * 8];
      bf4[j].q[1] = *(const uint4*)&Bbuf[col][16 + hf * 8];
    }
#pragma unroll
    for (int i = 0; i < 2; i++)
#pragma unroll
      for (int j = 0; j < 4; j++)
        acc[i][j] = wmma_bf16(af[i], bf4[j], acc[i][j]);
    __syncthreads();
  }

  // ---- epilogue: bias + store (D layout: M = v + 8*hf, N = ln) ----
#pragma unroll
  for (int i = 0; i < 2; i++) {
    const int orow0 = obase + wm * 32 + i * 16 + 8 * hf;
    float4 bv0 = *(const float4*)(bias + orow0);
    float4 bv1 = *(const float4*)(bias + orow0 + 4);
    float bvv[8] = {bv0.x, bv0.y, bv0.z, bv0.w, bv1.x, bv1.y, bv1.z, bv1.w};
#pragma unroll
    for (int j = 0; j < 4; j++) {
      const int nn = nbase + wn * 64 + j * 16 + ln;
      if constexpr (MODE == 0) {
        unsigned short* dst = Yb + (size_t)b * CC * NN;
#pragma unroll
        for (int v = 0; v < 8; v++)
          dst[(size_t)(orow0 + v) * NN + nn] = f2bf(acc[i][j][v] + bvv[v]);
      } else if constexpr (MODE == 1) {
        uint4 p; uint32_t* pw = (uint32_t*)&p;
#pragma unroll
        for (int v = 0; v < 4; v++) {
          pw[v] = (uint32_t)f2bf(acc[i][j][2 * v]     + bvv[2 * v]) |
                 ((uint32_t)f2bf(acc[i][j][2 * v + 1] + bvv[2 * v + 1]) << 16);
        }
        *(uint4*)(Yb + (size_t)b * CC * NN + (size_t)nn * CC + orow0) = p;
      } else {
        const size_t base = (size_t)b * CC * NN;
#pragma unroll
        for (int v = 0; v < 8; v++) {
          size_t idx = base + (size_t)(orow0 + v) * NN + nn;
          Yf[idx] = resid[idx] + acc[i][j][v] + bvv[v];
        }
      }
    }
  }
}

// ---------------------------------------------------------------------------
// Flash-style attention core: per WG a 64-query block, stream 64-key tiles.
// theta,phi: (B,N,C) bf16; g,z: (B,C,N) bf16. Operands (~50MB) live in the
// 192MB L2, so WMMA fragments are loaded straight from global (16B runs).
// 8 waves: phase A -> 16 S-tiles split 2/wave; phase C -> each wave owns a
// 64-channel slab of the 64x512 output accumulator (128 VGPRs/wave).
// ---------------------------------------------------------------------------
__global__ __launch_bounds__(256) void attn_kernel(
    const unsigned short* __restrict__ theta,
    const unsigned short* __restrict__ phi,
    const unsigned short* __restrict__ g,
    unsigned short* __restrict__ z)
{
  __shared__ __align__(16) float          S[64][65];
  __shared__ __align__(16) unsigned short P[64][64];   // bf16 probs (A-matrix source)
  __shared__ __align__(16) float rowM[64];
  __shared__ __align__(16) float rowL[64];
  __shared__ __align__(16) float fac[64];
  __shared__ float pmax[64][4];
  __shared__ float psum[64][4];

  const int tid  = threadIdx.x;
  const int wv   = tid >> 5, lane = tid & 31;
  const int hf   = lane >> 4, ln = lane & 15;
  const int b     = blockIdx.y;
  const int qbase = blockIdx.x * 64;
  const float scale = 0.04419417382415922f;  // 512^-0.5

  const unsigned short* thb = theta + (size_t)b * NN * CC;
  const unsigned short* phb = phi   + (size_t)b * NN * CC;
  const unsigned short* gb  = g     + (size_t)b * CC * NN;

  if (tid < 64) { rowM[tid] = -3.0e38f; rowL[tid] = 0.0f; }

  v8f zacc[4][4];
#pragma unroll
  for (int i = 0; i < 4; i++)
#pragma unroll
    for (int j = 0; j < 4; j++) zacc[i][j] = vzero();

  const int tqA = wv >> 1;        // wave's query tile for phase A (0..3)
  const int tmA = (wv & 1) * 2;   // wave's two key tiles for phase A
  const int cbase = wv * 64;      // wave's channel slab for phase C
  const unsigned short* arow = thb + (size_t)(qbase + tqA * 16 + ln) * CC;

  for (int mb = 0; mb < NN; mb += 64) {
    __syncthreads();

    // ---- phase A: S = theta^T @ phi over full C, 2 tiles/wave ----
    v8f s0 = vzero(), s1 = vzero();
    const unsigned short* brow0 = phb + (size_t)(mb + tmA * 16 + ln) * CC;
    const unsigned short* brow1 = brow0 + (size_t)16 * CC;
#pragma unroll 4
    for (int k = 0; k < CC; k += 32) {
      Frag a, b0, b1;
      const int o0 = k + hf * 8, o1 = k + 16 + hf * 8;
      a.q[0]  = *(const uint4*)(arow + o0);   a.q[1]  = *(const uint4*)(arow + o1);
      b0.q[0] = *(const uint4*)(brow0 + o0);  b0.q[1] = *(const uint4*)(brow0 + o1);
      b1.q[0] = *(const uint4*)(brow1 + o0);  b1.q[1] = *(const uint4*)(brow1 + o1);
      s0 = wmma_bf16(a, b0, s0);
      s1 = wmma_bf16(a, b1, s1);
    }
#pragma unroll
    for (int v = 0; v < 8; v++) {
      S[tqA * 16 + v + 8 * hf][tmA * 16 + ln]      = s0[v] * scale;
      S[tqA * 16 + v + 8 * hf][tmA * 16 + 16 + ln] = s1[v] * scale;
    }
    if (mb + 64 < NN) __builtin_prefetch(brow0 + (size_t)64 * CC, 0, 1);
    __syncthreads();

    // ---- phase B: online softmax over the 64-key tile ----
    const int r = tid >> 2, cg = (tid & 3) * 16;
    float mx = -3.0e38f;
#pragma unroll
    for (int j2 = 0; j2 < 16; j2++) mx = fmaxf(mx, S[r][cg + j2]);
    pmax[r][tid & 3] = mx;
    __syncthreads();
    const float oldM = rowM[r];
    const float newM = fmaxf(oldM, fmaxf(fmaxf(pmax[r][0], pmax[r][1]),
                                         fmaxf(pmax[r][2], pmax[r][3])));
    const float fsc = __expf(oldM - newM);
    float sum = 0.0f;
#pragma unroll
    for (int j2 = 0; j2 < 16; j2++) {
      float p = __expf(S[r][cg + j2] - newM);
      sum += p;
      P[r][cg + j2] = f2bf(p);
    }
    psum[r][tid & 3] = sum;
    __syncthreads();
    if ((tid & 3) == 0) {
      rowL[r] = rowL[r] * fsc + psum[r][0] + psum[r][1] + psum[r][2] + psum[r][3];
      rowM[r] = newM;
      fac[r]  = fsc;
    }
    __syncthreads();

    // ---- phase C: rescale running acc, then z += P @ g^T ----
#pragma unroll
    for (int i = 0; i < 4; i++) {
      float4 f0 = *(const float4*)&fac[i * 16 + 8 * hf];
      float4 f1 = *(const float4*)&fac[i * 16 + 8 * hf + 4];
      float fv[8] = {f0.x, f0.y, f0.z, f0.w, f1.x, f1.y, f1.z, f1.w};
#pragma unroll
      for (int j = 0; j < 4; j++)
#pragma unroll
        for (int v = 0; v < 8; v++) zacc[i][j][v] *= fv[v];
    }
#pragma unroll
    for (int k = 0; k < 64; k += 32) {
      Frag pa[4], gf[4];
#pragma unroll
      for (int i = 0; i < 4; i++) {
        pa[i].q[0] = *(const uint4*)&P[i * 16 + ln][k + hf * 8];
        pa[i].q[1] = *(const uint4*)&P[i * 16 + ln][k + 16 + hf * 8];
      }
#pragma unroll
      for (int j = 0; j < 4; j++) {
        const unsigned short* grow = gb + (size_t)(cbase + j * 16 + ln) * NN + mb;
        gf[j].q[0] = *(const uint4*)(grow + k + hf * 8);
        gf[j].q[1] = *(const uint4*)(grow + k + 16 + hf * 8);
      }
#pragma unroll
      for (int i = 0; i < 4; i++)
#pragma unroll
        for (int j = 0; j < 4; j++)
          zacc[i][j] = wmma_bf16(pa[i], gf[j], zacc[i][j]);
    }
  }

  __syncthreads();
  // ---- epilogue: z = acc / rowL, store bf16 (C,N), 16B per lane ----
#pragma unroll
  for (int i = 0; i < 4; i++) {
    float4 l0 = *(const float4*)&rowL[i * 16 + 8 * hf];
    float4 l1 = *(const float4*)&rowL[i * 16 + 8 * hf + 4];
    float inv[8] = {1.f / l0.x, 1.f / l0.y, 1.f / l0.z, 1.f / l0.w,
                    1.f / l1.x, 1.f / l1.y, 1.f / l1.z, 1.f / l1.w};
    const int q0 = qbase + i * 16 + 8 * hf;
#pragma unroll
    for (int j = 0; j < 4; j++) {
      const int c = cbase + j * 16 + ln;
      uint4 p; uint32_t* pw = (uint32_t*)&p;
#pragma unroll
      for (int v = 0; v < 4; v++) {
        pw[v] = (uint32_t)f2bf(zacc[i][j][2 * v]     * inv[2 * v]) |
               ((uint32_t)f2bf(zacc[i][j][2 * v + 1] * inv[2 * v + 1]) << 16);
      }
      *(uint4*)(z + (size_t)b * CC * NN + (size_t)c * NN + q0) = p;
    }
  }
}

// ---------------------------------------------------------------------------
extern "C" void kernel_launch(void* const* d_in, const int* in_sizes, int n_in,
                              void* d_out, int out_size, void* d_ws, size_t ws_size,
                              hipStream_t stream) {
  const float* x       = (const float*)d_in[0];
  const float* w_theta = (const float*)d_in[1];
  const float* b_theta = (const float*)d_in[2];
  const float* w_phi   = (const float*)d_in[3];
  const float* b_phi   = (const float*)d_in[4];
  const float* w_g     = (const float*)d_in[5];
  const float* b_g     = (const float*)d_in[6];
  const float* w_out   = (const float*)d_in[7];
  const float* b_out   = (const float*)d_in[8];
  float* out = (float*)d_out;

  // Workspace: 5 bf16 (B,C,N)/(B,N,C) tensors of 16.8MB each (~84MB total).
  const size_t SZ = (size_t)NB * CC * NN;          // elements per tensor
  unsigned short* gn  = (unsigned short*)d_ws;
  unsigned short* th  = gn + SZ;
  unsigned short* ph  = th + SZ;
  unsigned short* gm  = ph + SZ;
  unsigned short* zw  = gm + SZ;

  dim3 gemm_grid(NN / 128, CC / 128, NB);  // (32, 4, 4)
  dim3 blk(256);

  // gn = W_g @ x + b_g                         (fp32 in, bf16 (C,N) out)
  conv_gemm<0, 0><<<gemm_grid, blk, 0, stream>>>(w_g, b_g, x, gn, nullptr, nullptr);
  // theta = W_t @ gn + b_t                     (bf16 in, bf16 (N,C) out)
  conv_gemm<1, 1><<<gemm_grid, blk, 0, stream>>>(w_theta, b_theta, gn, th, nullptr, nullptr);
  // phi = W_p @ gn + b_p                       (bf16 in, bf16 (N,C) out)
  conv_gemm<1, 1><<<gemm_grid, blk, 0, stream>>>(w_phi, b_phi, gn, ph, nullptr, nullptr);
  // g2 = W_g @ gn + b_g                        (bf16 in, bf16 (C,N) out)
  conv_gemm<1, 0><<<gemm_grid, blk, 0, stream>>>(w_g, b_g, gn, gm, nullptr, nullptr);
  // z = softmax(theta^T phi * scale) applied to g2  (flash-style)
  attn_kernel<<<dim3(NN / 64, NB), blk, 0, stream>>>(th, ph, gm, zw);
  // out = x + W_o @ z + b_o                    (bf16 in, fp32 out + residual)
  conv_gemm<1, 2><<<gemm_grid, blk, 0, stream>>>(w_out, b_out, zw, nullptr, out, x);
}
